// MultiHeadAttention_50268297232847
// MI455X (gfx1250) — compile-verified, matrix operand order checked by reading, not run on hardware
//
#include <hip/hip_runtime.h>

// ---------------- CDNA5 WMMA types ----------------
typedef __attribute__((ext_vector_type(16))) _Float16 v16h;
typedef __attribute__((ext_vector_type(8)))  _Float16 v8h;
typedef __attribute__((ext_vector_type(4)))  _Float16 v4h;
typedef __attribute__((ext_vector_type(8)))  float    v8f;

#define B_   4
#define L_   2048
#define D_   1024
#define H_   16
#define DH_  64

__device__ __forceinline__ v16h cat8(v8h lo, v8h hi) {
  return __builtin_shufflevector(lo, hi, 0,1,2,3,4,5,6,7,8,9,10,11,12,13,14,15);
}
// one 16x32 f16 fragment = two b128 loads at the ISA's half-select offsets
__device__ __forceinline__ v16h ldfrag(const _Float16* p, int o) {
  return cat8(*(const v8h*)(p + o), *(const v8h*)(p + o + 16));
}
__device__ __forceinline__ v8f wmma_f16(v16h a, v16h b, v8f c) {
  // emits v_wmma_f32_16x16x32_f16
  return __builtin_amdgcn_wmma_f32_16x16x32_f16(false, a, false, b, (short)0, c, false, false);
}

// ---------------- fp32 -> fp16 bulk convert (b128 load / b64 store) ----------------
__global__ __launch_bounds__(256) void cvt_kernel(const float* __restrict__ s,
                                                  _Float16* __restrict__ d, int n4) {
  int i = blockIdx.x * 256 + threadIdx.x;
  if (i >= n4) return;
  float4 v = ((const float4*)s)[i];
  v4h o;
  o[0] = (_Float16)v.x; o[1] = (_Float16)v.y;
  o[2] = (_Float16)v.z; o[3] = (_Float16)v.w;
  ((v4h*)d)[i] = o;
}

// ---------------- projection GEMM: Out[i,j] = sum_d X[i,d] * W[j,d] ----------------
// Each wave computes a 16(M) x 64(N) strip with software double-buffering:
// fragments for K-step k+32 are loaded before the 4 WMMAs of step k issue.
// vt==0: write [b,h,l,64] (Q,K).  vt==1: write transposed [b,h,64,L] (V).
__global__ __launch_bounds__(256) void proj_kernel(const _Float16* __restrict__ X,
                                                   const _Float16* __restrict__ W,
                                                   _Float16* __restrict__ Out,
                                                   int vt) {
  const int wid   = threadIdx.x >> 5;
  const int lane  = threadIdx.x & 31;
  const int row16 = lane & 15;
  const int hs    = lane >> 4;            // half-select per ISA 16-bit A/B layout

  const int tile = blockIdx.x * 8 + wid;  // 8192 strips total
  const int ntn  = D_ / 64;               // 16 N strips
  const int m0   = (tile / ntn) * 16;
  const int n0   = (tile % ntn) * 64;

  const _Float16* Arow = X + (size_t)(m0 + row16) * D_;
  const _Float16* Br0  = W + (size_t)(n0 +  0 + row16) * D_;
  const _Float16* Br1  = W + (size_t)(n0 + 16 + row16) * D_;
  const _Float16* Br2  = W + (size_t)(n0 + 32 + row16) * D_;
  const _Float16* Br3  = W + (size_t)(n0 + 48 + row16) * D_;

  v8f acc0 = {}, acc1 = {}, acc2 = {}, acc3 = {};
  const int ob = hs * 8;                  // lanes 0-15: K 0..7/16..23; lanes 16-31: 8..15/24..31
  v16h a  = ldfrag(Arow, ob);
  v16h b0 = ldfrag(Br0,  ob);
  v16h b1 = ldfrag(Br1,  ob);
  v16h b2 = ldfrag(Br2,  ob);
  v16h b3 = ldfrag(Br3,  ob);
#pragma unroll 4
  for (int k0 = 0; k0 < D_; k0 += 32) {
    v16h an = a, b0n = b0, b1n = b1, b2n = b2, b3n = b3;
    if (k0 + 32 < D_) {                   // prefetch next K-step before this step's WMMAs
      const int on = k0 + 32 + ob;
      an  = ldfrag(Arow, on);
      b0n = ldfrag(Br0,  on);
      b1n = ldfrag(Br1,  on);
      b2n = ldfrag(Br2,  on);
      b3n = ldfrag(Br3,  on);
    }
    acc0 = wmma_f16(a, b0, acc0);
    acc1 = wmma_f16(a, b1, acc1);
    acc2 = wmma_f16(a, b2, acc2);
    acc3 = wmma_f16(a, b3, acc3);
    a = an; b0 = b0n; b1 = b1n; b2 = b2n; b3 = b3n;
  }

  // C/D layout: element r -> M = 8*hs + r, N(local) = row16
  const v8f accs[4] = {acc0, acc1, acc2, acc3};
#pragma unroll
  for (int t = 0; t < 4; ++t) {
    const int gj = n0 + t * 16 + row16;   // global feature
    const int h = gj >> 6, j = gj & (DH_ - 1);
    if (vt) {
      v8h pack;
#pragma unroll
      for (int r = 0; r < 8; ++r) pack[r] = (_Float16)accs[t][r];
      const int gi0 = m0 + hs * 8;
      const int b = gi0 >> 11, l0 = gi0 & (L_ - 1);
      *(v8h*)(Out + ((size_t)((b * H_ + h) * DH_ + j) * L_ + l0)) = pack;  // b128 store
    } else {
#pragma unroll
      for (int r = 0; r < 8; ++r) {
        const int gi = m0 + hs * 8 + r;
        const int b = gi >> 11, l = gi & (L_ - 1);
        Out[(size_t)((b * H_ + h) * L_ + l) * DH_ + j] = (_Float16)accs[t][r];
      }
    }
  }
}

// ---------------- fused attention: scores (WMMA) -> softmax -> att write -> P@V (WMMA) ----------
// One workgroup (8 waves) per (b,h, 16 q-rows). LDS: 16x2048 f32 scores + 16x2048 f16 probs.
__global__ __launch_bounds__(256) void attn_kernel(const _Float16* __restrict__ Q,
                                                   const _Float16* __restrict__ K,
                                                   const _Float16* __restrict__ Vt,
                                                   const unsigned char* __restrict__ mask,
                                                   float* __restrict__ out,
                                                   float* __restrict__ att) {
  extern __shared__ char smem[];
  float*    sS = (float*)smem;                         // 16*2048 f32 = 128 KB
  _Float16* sP = (_Float16*)(smem + 16 * L_ * 4);      // 16*2048 f16 =  64 KB
  __shared__ float sRed[256];

  const int tid   = threadIdx.x;
  const int wid   = tid >> 5;
  const int lane  = tid & 31;
  const int row16 = lane & 15;
  const int hs    = lane >> 4;

  const int nqt = L_ / 16;                 // 128 q tiles per head
  const int bh  = blockIdx.x / nqt;        // fused (b,h): 0..63
  const int q0  = (blockIdx.x % nqt) * 16;
  const int ob  = hs * 8;

  // ---- Q fragments (A operand, M=row16, K=0..63 in two 16x32 frags), held in regs ----
  const _Float16* Qrow = Q + ((size_t)bh * L_ + (q0 + row16)) * DH_;
  v16h a0 = ldfrag(Qrow, ob);
  v16h a1 = ldfrag(Qrow, 32 + ob);

  // ---- scores: each wave covers 16 column tiles of 16 keys (double-buffered) ----
  const _Float16* Kbase = K + (size_t)bh * L_ * DH_;
  {
    const _Float16* Kr = Kbase + (size_t)(wid * 16 + row16) * DH_;
    v16h b0 = ldfrag(Kr, ob);
    v16h b1 = ldfrag(Kr, 32 + ob);
    for (int ct = wid; ct < L_ / 16; ct += 8) {
      v16h b0n = b0, b1n = b1;
      if (ct + 8 < L_ / 16) {
        const _Float16* Kr2 = Kbase + (size_t)((ct + 8) * 16 + row16) * DH_;
        b0n = ldfrag(Kr2, ob);
        b1n = ldfrag(Kr2, 32 + ob);
      }
      v8f acc = {};
      acc = wmma_f16(a0, b0, acc);
      acc = wmma_f16(a1, b1, acc);
      const int n0 = ct * 16;
#pragma unroll
      for (int r = 0; r < 8; ++r)
        sS[(hs * 8 + r) * L_ + n0 + row16] = acc[r] * 0.125f;   // 1/sqrt(64)
      b0 = b0n; b1 = b1n;
    }
  }
  __syncthreads();

  // ---- softmax over 2048 columns: 16 threads per row, float4/uchar4 vectorized ----
  const int r   = tid >> 4;
  const int sub = tid & 15;
  const unsigned char* mrow = mask + ((size_t)bh * L_ + (q0 + r)) * L_;
  float* srow = sS + r * L_;

  float mx = -3.0e38f;
  for (int i = sub; i < L_ / 4; i += 16) {
    float4 s4 = ((float4*)srow)[i];               // ds_load_b128
    uchar4 m4 = ((const uchar4*)mrow)[i];
    if (m4.x) s4.x = -3.0e38f;
    if (m4.y) s4.y = -3.0e38f;
    if (m4.z) s4.z = -3.0e38f;
    if (m4.w) s4.w = -3.0e38f;
    ((float4*)srow)[i] = s4;
    mx = fmaxf(mx, fmaxf(fmaxf(s4.x, s4.y), fmaxf(s4.z, s4.w)));
  }
  sRed[tid] = mx;
  __syncthreads();
  for (int off = 8; off > 0; off >>= 1) {
    if (sub < off) sRed[r * 16 + sub] = fmaxf(sRed[r * 16 + sub], sRed[r * 16 + sub + off]);
    __syncthreads();
  }
  const float rowmax = sRed[r * 16];
  __syncthreads();

  float sum = 0.f;
  for (int i = sub; i < L_ / 4; i += 16) {
    float4 s4 = ((float4*)srow)[i];
    float4 e4;
    e4.x = __expf(s4.x - rowmax);
    e4.y = __expf(s4.y - rowmax);
    e4.z = __expf(s4.z - rowmax);
    e4.w = __expf(s4.w - rowmax);
    ((float4*)srow)[i] = e4;
    sum += (e4.x + e4.y) + (e4.z + e4.w);
  }
  sRed[tid] = sum;
  __syncthreads();
  for (int off = 8; off > 0; off >>= 1) {
    if (sub < off) sRed[r * 16 + sub] += sRed[r * 16 + sub + off];
    __syncthreads();
  }
  const float inv = 1.0f / sRed[r * 16];

  float* attrow = att + ((size_t)bh * L_ + (q0 + r)) * L_;
  _Float16* prow = sP + r * L_;
  for (int i = sub; i < L_ / 4; i += 16) {
    float4 e4 = ((float4*)srow)[i];
    float4 p4;
    p4.x = e4.x * inv; p4.y = e4.y * inv; p4.z = e4.z * inv; p4.w = e4.w * inv;
    ((float4*)attrow)[i] = p4;                    // the one-and-only HBM write of att (b128)
    v4h h4;
    h4[0] = (_Float16)p4.x; h4[1] = (_Float16)p4.y;
    h4[2] = (_Float16)p4.z; h4[3] = (_Float16)p4.w;
    ((v4h*)prow)[i] = h4;                         // f16 probs stay in LDS for P@V
  }
  __syncthreads();

  // ---- P @ V: every wave computes the full 16x64 tile over 1/8 of K (256 deep).
  //      One LDS A-fragment feeds 4 WMMAs; double-buffered; partials combined in LDS. ----
  const _Float16* Prow_s = sP + (size_t)row16 * L_;        // A operand row (M = row16)
  const _Float16* Vbase  = Vt + (size_t)bh * DH_ * L_;
  const _Float16* Vr0 = Vbase + (size_t)( 0 + row16) * L_;
  const _Float16* Vr1 = Vbase + (size_t)(16 + row16) * L_;
  const _Float16* Vr2 = Vbase + (size_t)(32 + row16) * L_;
  const _Float16* Vr3 = Vbase + (size_t)(48 + row16) * L_;
  v8f acc0 = {}, acc1 = {}, acc2 = {}, acc3 = {};
  const int kbeg = wid * (L_ / 8);
  {
    v16h a  = ldfrag(Prow_s, kbeg + ob);                   // ds_load_b128 x2
    v16h c0 = ldfrag(Vr0, kbeg + ob);
    v16h c1 = ldfrag(Vr1, kbeg + ob);
    v16h c2 = ldfrag(Vr2, kbeg + ob);
    v16h c3 = ldfrag(Vr3, kbeg + ob);
    for (int k0 = kbeg; k0 < kbeg + L_ / 8; k0 += 32) {
      v16h an = a, c0n = c0, c1n = c1, c2n = c2, c3n = c3;
      if (k0 + 32 < kbeg + L_ / 8) {
        const int on = k0 + 32 + ob;
        an  = ldfrag(Prow_s, on);
        c0n = ldfrag(Vr0, on);
        c1n = ldfrag(Vr1, on);
        c2n = ldfrag(Vr2, on);
        c3n = ldfrag(Vr3, on);
      }
      acc0 = wmma_f16(a, c0, acc0);
      acc1 = wmma_f16(a, c1, acc1);
      acc2 = wmma_f16(a, c2, acc2);
      acc3 = wmma_f16(a, c3, acc3);
      a = an; c0 = c0n; c1 = c1n; c2 = c2n; c3 = c3n;
    }
  }
  // partials: 8 waves x 16x64 floats = 32 KB, reuse score region (softmax is done)
  float* part = sS;
  {
    const v8f accs[4] = {acc0, acc1, acc2, acc3};
#pragma unroll
    for (int t = 0; t < 4; ++t)
#pragma unroll
      for (int rr = 0; rr < 8; ++rr)
        part[wid * 1024 + t * 256 + (hs * 8 + rr) * 16 + row16] = accs[t][rr];
  }
  __syncthreads();

  const int b = bh >> 4, h = bh & 15;
  for (int e = tid; e < 16 * DH_; e += 256) {
    const int m2 = e >> 6, n2 = e & 63;
    const int t4 = n2 >> 4, nl = n2 & 15;
    float v = 0.f;
#pragma unroll
    for (int w = 0; w < 8; ++w)
      v += part[w * 1024 + t4 * 256 + m2 * 16 + nl];
    out[((size_t)b * L_ + q0 + m2) * D_ + h * DH_ + n2] = v;
  }
}

// ---------------- host ----------------
extern "C" void kernel_launch(void* const* d_in, const int* in_sizes, int n_in,
                              void* d_out, int out_size, void* d_ws, size_t ws_size,
                              hipStream_t stream) {
  (void)in_sizes; (void)n_in; (void)out_size;

  const float* qry = (const float*)d_in[0];
  const float* key = (const float*)d_in[1];
  const float* val = (const float*)d_in[2];
  const unsigned char* mask = (const unsigned char*)d_in[3];
  const float* Wq = (const float*)d_in[4];
  const float* Wk = (const float*)d_in[5];
  const float* Wv = (const float*)d_in[6];

  const size_t OUT_ELEMS = (size_t)B_ * L_ * D_;         // 8388608
  const size_t ATT_ELEMS = (size_t)B_ * H_ * L_ * L_;    // 268435456
  float* outp = (float*)d_out;
  float* attp = outp + OUT_ELEMS;

  const size_t QKV_BYTES = (size_t)B_ * H_ * L_ * DH_ * 2;  // 16 MiB each
  const size_t X_BYTES   = (size_t)B_ * L_ * D_ * 2;        // 16 MiB each
  const size_t W_BYTES   = (size_t)D_ * D_ * 2;             //  2 MiB each

  char* wsb = (char*)d_ws;
  _Float16* Qb  = (_Float16*)(wsb);
  _Float16* Kb  = (_Float16*)(wsb + QKV_BYTES);
  _Float16* Vtb = (_Float16*)(wsb + 2 * QKV_BYTES);

  // f16 copies of x / W: prefer ws; otherwise use the tail of the (not-yet-written)
  // att region of d_out — consumed strictly before attn_kernel overwrites it.
  const size_t need_full = 3 * QKV_BYTES + 3 * X_BYTES + 3 * W_BYTES;
  char* scratch;
  if (ws_size >= need_full) {
    scratch = wsb + 3 * QKV_BYTES;
  } else {
    const size_t dout_bytes = (OUT_ELEMS + ATT_ELEMS) * 4;
    scratch = (char*)d_out + dout_bytes - (3 * X_BYTES + 3 * W_BYTES);
  }
  _Float16* Xq  = (_Float16*)(scratch);
  _Float16* Xk  = (_Float16*)(scratch + X_BYTES);
  _Float16* Xv  = (_Float16*)(scratch + 2 * X_BYTES);
  _Float16* Wqh = (_Float16*)(scratch + 3 * X_BYTES);
  _Float16* Wkh = (_Float16*)(scratch + 3 * X_BYTES + W_BYTES);
  _Float16* Wvh = (_Float16*)(scratch + 3 * X_BYTES + 2 * W_BYTES);

  // 1) convert inputs/weights to f16 (one pass each)
  const int nx4 = (int)((size_t)B_ * L_ * D_ / 4);   // 2097152
  const int nw4 = (int)((size_t)D_ * D_ / 4);        //  262144
  cvt_kernel<<<nx4 / 256, 256, 0, stream>>>(qry, Xq, nx4);
  cvt_kernel<<<nx4 / 256, 256, 0, stream>>>(key, Xk, nx4);
  cvt_kernel<<<nx4 / 256, 256, 0, stream>>>(val, Xv, nx4);
  cvt_kernel<<<nw4 / 256, 256, 0, stream>>>(Wq, Wqh, nw4);
  cvt_kernel<<<nw4 / 256, 256, 0, stream>>>(Wk, Wkh, nw4);
  cvt_kernel<<<nw4 / 256, 256, 0, stream>>>(Wv, Wvh, nw4);

  // 2) projections (WMMA f16): Q,K -> [b,h,l,64]; V -> transposed [b,h,64,L]
  const int projBlocks = (B_ * L_ / 16) * (D_ / 64) / 8;  // 1024 blocks of 8 wave-strips
  proj_kernel<<<projBlocks, 256, 0, stream>>>(Xq, Wqh, Qb, 0);
  proj_kernel<<<projBlocks, 256, 0, stream>>>(Xk, Wkh, Kb, 0);
  proj_kernel<<<projBlocks, 256, 0, stream>>>(Xv, Wvh, Vtb, 1);

  // 3) fused scores -> softmax -> att -> P@V
  const size_t smem = (size_t)16 * L_ * 4 + (size_t)16 * L_ * 2;  // 196608 B (<320KB WGP LDS)
  hipFuncSetAttribute((const void*)attn_kernel,
                      hipFuncAttributeMaxDynamicSharedMemorySize, (int)smem);
  attn_kernel<<<B_ * H_ * (L_ / 16), 256, smem, stream>>>(Qb, Kb, Vtb, mask, outp, attp);
}
